// Model_15470472200626
// MI455X (gfx1250) — matrix operand order repaired and checked by reading, not compiled, not run on hardware
//
#include <hip/hip_runtime.h>
#include <math.h>

// MI455X / gfx1250: wave32, WMMA f32_16x16x32_f16 path.
typedef __attribute__((ext_vector_type(16))) _Float16 v16h;
typedef __attribute__((ext_vector_type(8)))  _Float16 v8h;
typedef __attribute__((ext_vector_type(8)))  float    v8f;

#define B_N 2048
#define J_N 24
#define H_N 24
#define ROWS (B_N * J_N)       // 49152 flat (batch, joint) rows
#define STATE_N (ROWS * H_N)   // 1179648 floats per state tensor

// SMPL kinematic tree: neighbor lists (parent + children), -1 padded, and 1/deg
__constant__ int c_nbr[24][4] = {
  { 1, 2, 3,-1},{ 0, 4,-1,-1},{ 0, 5,-1,-1},{ 0, 6,-1,-1},
  { 1, 7,-1,-1},{ 2, 8,-1,-1},{ 3, 9,-1,-1},{ 4,10,-1,-1},
  { 5,11,-1,-1},{ 6,12,13,14},{ 7,-1,-1,-1},{ 8,-1,-1,-1},
  { 9,15,-1,-1},{ 9,16,-1,-1},{ 9,17,-1,-1},{12,-1,-1,-1},
  {13,18,-1,-1},{14,19,-1,-1},{16,20,-1,-1},{17,21,-1,-1},
  {18,22,-1,-1},{19,23,-1,-1},{20,-1,-1,-1},{21,-1,-1,-1}
};
__constant__ float c_invdeg[24] = {
  0.25f, 1.f/3, 1.f/3, 1.f/3, 1.f/3, 1.f/3, 1.f/3, 1.f/3,
  1.f/3, 0.2f,  0.5f,  0.5f,  1.f/3, 1.f/3, 1.f/3, 0.5f,
  1.f/3, 1.f/3, 1.f/3, 1.f/3, 1.f/3, 1.f/3, 0.5f,  0.5f
};

__device__ __forceinline__ float sigmoidf_(float x) { return 1.0f / (1.0f + __expf(-x)); }

// ---------------------------------------------------------------------------
// Encoder: out[b,j,:] = relu(x[b,frame0,j,:] @ We^T + be)   (K=3, pure VALU)
// ---------------------------------------------------------------------------
__global__ void enc_kernel(const float* __restrict__ seq, const float* __restrict__ We,
                           const float* __restrict__ be, float* __restrict__ out) {
  int idx = blockIdx.x * blockDim.x + threadIdx.x;
  if (idx >= STATE_N) return;
  int hh = idx % H_N;
  int bj = idx / H_N;
  int b = bj / J_N, j = bj % J_N;
  const float* x = seq + ((size_t)b * 24 /*S*/ + 0) * J_N * 3 + (size_t)j * 3; // frame 0
  float v = be[hh] + x[0] * We[hh*3+0] + x[1] * We[hh*3+1] + x[2] * We[hh*3+2];
  out[idx] = fmaxf(v, 0.0f);
}

// ---------------------------------------------------------------------------
// Prep: pack weights in *WMMA B-fragment order* so GEMM kernels load each
// fragment as one contiguous 32-byte v16h.
//  fragment(ks,ct,lane)[e] = W[k = ks*32 + (lane>>4)*16 + e][n = ct*16 + (lane&15)]
// Wfrag : cell GEMM, logical W_big[96K x 96N] (gate-concat, transposed)
// Wdfrag: out projection, logical Wd^T [576K x 80N] (n>=72 zero-padded)
// bias96 = bxl + bhl + bg (folded)
// ---------------------------------------------------------------------------
__global__ void prep_kernel(const float* __restrict__ Wxl, const float* __restrict__ Wxr,
                            const float* __restrict__ Whl, const float* __restrict__ Whr,
                            const float* __restrict__ bxl, const float* __restrict__ bhl,
                            const float* __restrict__ bg,  const float* __restrict__ Wd,
                            _Float16* __restrict__ Wfrag, _Float16* __restrict__ Wdfrag,
                            float* __restrict__ bias96) {
  for (int i = blockIdx.x * blockDim.x + threadIdx.x; i < 576 * 80; i += gridDim.x * blockDim.x) {
    if (i < 96) bias96[i] = bxl[i] + bhl[i] + bg[i];
    if (i < 96 * 96) {  // cell W fragments: 3 ks x 6 ct x 32 lanes x 16
      int e    = i & 15;
      int lane = (i >> 4) & 31;
      int ct   = (i >> 9) % 6;
      int ks   = i / (512 * 6);
      int n = ct * 16 + (lane & 15);
      int k = ks * 32 + (lane >> 4) * 16 + e;
      int g = n / 24, nn = n % 24;
      float v;
      if (k < 24)      v = Wxl[(g*24 + nn)*24 + k];
      else if (k < 48) v = Wxr[(g*24 + nn)*24 + (k-24)];
      else if (k < 72) v = Whl[(g*24 + nn)*24 + (k-48)];
      else             v = Whr[(g*24 + nn)*24 + (k-72)];
      Wfrag[i] = (_Float16)v;
    }
    {   // out-proj Wd fragments: 18 ks x 5 ct x 32 lanes x 16
      int e    = i & 15;
      int lane = (i >> 4) & 31;
      int ct   = (i >> 9) % 5;
      int ks   = i / (512 * 5);
      int n = ct * 16 + (lane & 15);
      int k = ks * 32 + (lane >> 4) * 16 + e;
      Wdfrag[i] = (_Float16)((n < 72) ? Wd[(size_t)n*576 + k] : 0.0f);
    }
  }
}

// ---------------------------------------------------------------------------
// Fused GC-LSTM cell step.
// Block = 192 threads (6 waves), handles 4 batches (96 rows of M).
//   phase 1: build M = [aggX | X | aggH | h] fp16 in LDS (vectorized, 8-col chunks)
//   phase 2: each wave owns one 16-row tile: 6 N-tiles x 3 K-steps WMMA;
//            A-fragments from LDS (2x ds_load_b128), B-fragments straight
//            from global fragment-ordered W (L2 resident, used once per wave).
//   phase 3: gate panel round-trip via LDS (col-major, vector stores) ->
//            sigmoid/tanh + peepholes, in-place h/c update, write o.
// ---------------------------------------------------------------------------
#define MP 104   // M LDS pitch in halves (208 B rows -> 16B aligned chunks)
#define GP 104   // gate panel pitch in floats, col-major (416 B -> 32B aligned)

union CellShm {
  _Float16 M[96 * MP];   // 19968 B  (phases 1-2)
  float    GT[96 * GP];  // 39936 B  (phase 3; GT[col][row])
};

__global__ __launch_bounds__(192) void cell_kernel(
    const float* __restrict__ X, float* __restrict__ h, float* __restrict__ c,
    float* __restrict__ out_o,
    const _Float16* __restrict__ Wfrag, const float* __restrict__ bias96,
    const float* __restrict__ wg) {
  __shared__ CellShm sh;
  __shared__ float sBias[96];
  __shared__ float sWg[72];
  const int tid  = threadIdx.x;
  const int row0 = blockIdx.x * 96;   // flat (b*24+j) base (4 batches)

  if (tid < 96) sBias[tid] = bias96[tid];
  if (tid < 72) sWg[tid]   = wg[tid];

  // ---- phase 1: build M panel, 8-column vector chunks (12 per row)
  for (int ch = tid; ch < 96 * 12; ch += 192) {
    int r    = ch / 12;
    int cseg = ch % 12;
    int cc0  = cseg * 8;
    int grow  = row0 + r;
    int j     = r % 24;
    int bbase = grow - j;   // joint-0 row of this batch
    v8f s;
    if (cseg < 3) {                        // aggX
      s = *(const v8f*)(X + (size_t)grow * 24 + cc0);
      #pragma unroll
      for (int q = 0; q < 4; ++q) {
        int nb = c_nbr[j][q];
        if (nb >= 0) s += *(const v8f*)(X + (size_t)(bbase + nb) * 24 + cc0);
      }
      s *= c_invdeg[j];
    } else if (cseg < 6) {                 // X
      s = *(const v8f*)(X + (size_t)grow * 24 + (cc0 - 24));
    } else if (cseg < 9) {                 // aggH
      s = *(const v8f*)(h + (size_t)grow * 24 + (cc0 - 48));
      #pragma unroll
      for (int q = 0; q < 4; ++q) {
        int nb = c_nbr[j][q];
        if (nb >= 0) s += *(const v8f*)(h + (size_t)(bbase + nb) * 24 + (cc0 - 48));
      }
      s *= c_invdeg[j];
    } else {                               // h
      s = *(const v8f*)(h + (size_t)grow * 24 + (cc0 - 72));
    }
    v8h hv;
    #pragma unroll
    for (int e = 0; e < 8; ++e) hv[e] = (_Float16)s[e];
    *(v8h*)(&sh.M[r * MP + cc0]) = hv;     // ds_store_b128
  }
  __syncthreads();

  // ---- phase 2: WMMA  G[96x96] = M[96x96] @ W[96x96]
  const int w      = tid >> 5;   // wave id == row tile (0..5)
  const int lane   = tid & 31;
  const int ln     = lane & 15;
  const int laneHi = lane >> 4;  // which K-half this lane holds

  // A fragments: two contiguous 8-half runs per lane (adjacent: K = 16*half..+15, contiguous)
  v16h afr[3];
  #pragma unroll
  for (int ks = 0; ks < 3; ++ks) {
    const _Float16* mrow = &sh.M[(w * 16 + ln) * MP + ks * 32];
    v8h lo = *(const v8h*)(mrow + laneHi * 16);       // K = 16*half..+7
    v8h hi = *(const v8h*)(mrow + laneHi * 16 + 8);   // K = 16*half+8..+15
    afr[ks] = __builtin_shufflevector(lo, hi, 0,1,2,3,4,5,6,7,8,9,10,11,12,13,14,15);
  }
  const v16h* Wf = (const v16h*)Wfrag;
  v8f acc[6];
  #pragma unroll
  for (int ct = 0; ct < 6; ++ct) {
    v8f a = {0.f, 0.f, 0.f, 0.f, 0.f, 0.f, 0.f, 0.f};
    #pragma unroll
    for (int ks = 0; ks < 3; ++ks) {
      v16h bfr = Wf[(ks * 6 + ct) * 32 + lane];       // 32B contiguous global
      a = __builtin_amdgcn_wmma_f32_16x16x32_f16(false, afr[ks], false, bfr,
                                                 (short)0, a, false, false);
    }
    acc[ct] = a;
  }
  __syncthreads();   // all waves done reading M; LDS becomes gate panel

  // ---- scatter D tiles col-major: lane owns col n, 8 consecutive rows
  #pragma unroll
  for (int ct = 0; ct < 6; ++ct) {
    int n = ct * 16 + ln;
    *(v8f*)(&sh.GT[n * GP + w * 16 + laneHi * 8]) = acc[ct];  // 32B LDS store
  }
  __syncthreads();

  // ---- phase 3: elementwise LSTM (gates at cols h, 24+h, 48+h, 72+h)
  for (int it = 0; it < 12; ++it) {
    int e  = lane + it * 32;       // 0..383 over 16 rows x 24 h
    int r  = e / 24;
    int hh = e % 24;
    int rl = w * 16 + r;           // row within block panel
    int grow = row0 + rl;
    size_t idx = (size_t)grow * 24 + hh;
    float Gi = sh.GT[(hh)      * GP + rl];
    float Gf = sh.GT[(24 + hh) * GP + rl];
    float Gc = sh.GT[(48 + hh) * GP + rl];
    float Go = sh.GT[(72 + hh) * GP + rl];
    float cold = c[idx];
    float iv = sigmoidf_(Gi + sWg[hh]      * cold + sBias[hh]);
    float fv = sigmoidf_(Gf + sWg[24 + hh] * cold + sBias[24 + hh]);
    float cn = fv * cold + iv * tanhf(Gc + sBias[48 + hh]);
    float ov = sigmoidf_(Go + sWg[48 + hh] * cn + sBias[72 + hh]);
    out_o[idx] = ov;               // cell output (fed to next step)
    h[idx]     = ov * tanhf(cn);
    c[idx]     = cn;
  }
}

// ---------------------------------------------------------------------------
// Output projection for decode step t: dout[b,t,:72] = act[b,:576] @ Wd^T + bd
// WMMA GEMM: rows=2048, K=576 (18 k-steps), N=80 (5 tiles, 72 valid).
// ---------------------------------------------------------------------------
__global__ __launch_bounds__(128) void outproj_kernel(
    const float* __restrict__ act, const _Float16* __restrict__ Wdfrag,
    const float* __restrict__ bd, float* __restrict__ dout, int t) {
  const int lane   = threadIdx.x & 31;
  const int wv     = threadIdx.x >> 5;
  const int gw     = blockIdx.x * 4 + wv;   // row tile 0..127
  const int ln     = lane & 15;
  const int laneHi = lane >> 4;
  const int brow   = gw * 16 + ln;          // batch row for A fragment

  const v16h* Wf = (const v16h*)Wdfrag;
  v8f acc[5] = {};
  for (int ks = 0; ks < 18; ++ks) {
    const float* arow = act + (size_t)brow * 576 + ks * 32;
    v8f lo = *(const v8f*)(arow + laneHi * 16);
    v8f hi = *(const v8f*)(arow + laneHi * 16 + 8);
    v16h afr;
    #pragma unroll
    for (int e = 0; e < 8; ++e) { afr[e] = (_Float16)lo[e]; afr[8 + e] = (_Float16)hi[e]; }
    #pragma unroll
    for (int ct = 0; ct < 5; ++ct) {
      v16h bfr = Wf[(ks * 5 + ct) * 32 + lane];       // 32B contiguous global
      acc[ct] = __builtin_amdgcn_wmma_f32_16x16x32_f16(false, afr, false, bfr,
                                                       (short)0, acc[ct], false, false);
    }
  }
  #pragma unroll
  for (int ct = 0; ct < 5; ++ct) {
    int n = ct * 16 + ln;
    if (n < 72) {
      float bias = bd[n];
      #pragma unroll
      for (int i2 = 0; i2 < 8; ++i2) {
        int b = gw * 16 + laneHi * 8 + i2;
        dout[((size_t)b * 24 + t) * 72 + n] = acc[ct][i2] + bias;
      }
    }
  }
}

// ---------------------------------------------------------------------------
extern "C" void kernel_launch(void* const* d_in, const int* in_sizes, int n_in,
                              void* d_out, int out_size, void* d_ws, size_t ws_size,
                              hipStream_t stream) {
  const float* src = (const float*)d_in[0];
  const float* tgt = (const float*)d_in[1];
  const float* We  = (const float*)d_in[2];
  const float* be  = (const float*)d_in[3];
  const float* Wxl = (const float*)d_in[4];
  const float* bxl = (const float*)d_in[5];
  const float* Wxr = (const float*)d_in[6];
  const float* Whl = (const float*)d_in[7];
  const float* bhl = (const float*)d_in[8];
  const float* Whr = (const float*)d_in[9];
  const float* wg  = (const float*)d_in[10];
  const float* bg  = (const float*)d_in[11];
  const float* Wd  = (const float*)d_in[12];
  const float* bd  = (const float*)d_in[13];
  float* out = (float*)d_out;

  float* ws = (float*)d_ws;
  const size_t N = STATE_N;
  float* hbuf = ws;            // [B*J*H]
  float* cbuf = ws + N;
  float* x0   = ws + 2 * N;    // ping-pong activation buffers
  float* x1   = ws + 3 * N;
  float* xt   = ws + 4 * N;    // enc(tgt[:,0])
  _Float16* Wfrag  = (_Float16*)(ws + 5 * N);   // 96x96 fp16 (fragment order)
  _Float16* Wdfrag = Wfrag + 96 * 96;           // 576x80 fp16 (fragment order)
  float* bias96    = (float*)(Wdfrag + 576 * 80);

  hipMemsetAsync(hbuf, 0, 2 * N * sizeof(float), stream);  // h = c = 0
  prep_kernel<<<180, 256, 0, stream>>>(Wxl, Wxr, Whl, Whr, bxl, bhl, bg, Wd,
                                       Wfrag, Wdfrag, bias96);
  enc_kernel<<<(int)((N + 255) / 256), 256, 0, stream>>>(src, We, be, x0);
  enc_kernel<<<(int)((N + 255) / 256), 256, 0, stream>>>(tgt, We, be, xt);

  // warm-up: 24 dependent cell steps (ping-pong x0/x1, h/c in place)
  float* xin  = x0;
  float* xout = x1;
  for (int s = 0; s < 24; ++s) {
    cell_kernel<<<ROWS / 96, 192, 0, stream>>>(xin, hbuf, cbuf, xout,
                                               Wfrag, bias96, wg);
    float* tmp = xin; xin = xout; xout = tmp;
  }

  // decode: first input enc(tgt[:,0]), then feed previous output; project each
  const float* dxin = xt;
  float* dxout = x0;
  for (int t = 0; t < 24; ++t) {
    cell_kernel<<<ROWS / 96, 192, 0, stream>>>(dxin, hbuf, cbuf, dxout,
                                               Wfrag, bias96, wg);
    outproj_kernel<<<32, 128, 0, stream>>>(dxout, Wdfrag, bd, out, t);
    dxin  = dxout;
    dxout = (dxout == x0) ? x1 : x0;
  }
}
